// CrossAttnHead_26663156973632
// MI455X (gfx1250) — compile-verified, matrix operand order checked by reading, not run on hardware
//
#include <hip/hip_runtime.h>
#include <hip/hip_bf16.h>

// ---------------------------------------------------------------------------
// CrossAttnHead for MI455X (gfx1250): bf16 WMMA, fp32 accumulate.
// All GEMM operands arranged so each 16-bf16 WMMA operand = two b128 loads.
// Shapes: B=8 H=32 h=64 N_TAR=256 M=P=32 N_OBS=512 NH=8 DH=32 DT=256
// ---------------------------------------------------------------------------

typedef __attribute__((ext_vector_type(16))) __bf16 v16bf;
typedef __attribute__((ext_vector_type(8)))  float  v8f;

union ABop {                    // one WMMA 16-bit operand (8 VGPRs)
    v16bf        v;
    __bf16       e[16];
    uint4        q[2];
};
union F16x {                    // 16 fp32 staging values (A operand source)
    float4 f4[4];
    float  f[16];
};
union PK8 {                     // 8 bf16 packed for b128 store
    uint4  q;
    __bf16 e[8];
};
union PK4 {                     // 4 bf16 packed for b64 store
    uint2  d;
    __bf16 e[4];
};

__device__ __forceinline__ __bf16 f2bf(float f) { return (__bf16)f; }

__device__ __forceinline__ v8f wmma_bf16(const v16bf& a, const v16bf& b, const v8f& c) {
    return __builtin_amdgcn_wmma_f32_16x16x32_bf16(false, a, false, b, (short)0, c,
                                                   false, false);
}

// Operand element->K map (16-bit 16x32, wave32):
//   lanes 0-15 : e0..7 -> K 0..7,  e8..15 -> K 16..23
//   lanes 16-31: e0..7 -> K 8..15, e8..15 -> K 24..31
// => per lane: two contiguous 8-element runs at K = 8*hi and K = 8*hi+16.
__device__ __forceinline__ void load_a_f32(ABop& a, const float* base) {
    F16x s;
    const float4* p = (const float4*)base;
    s.f4[0] = p[0]; s.f4[1] = p[1];      // K +0..7
    s.f4[2] = p[4]; s.f4[3] = p[5];      // K +16..23
#pragma unroll
    for (int e = 0; e < 16; ++e) a.e[e] = f2bf(s.f[e]);
}
__device__ __forceinline__ void load_bf16_runs(ABop& a, const __bf16* base) {
    const uint4* p = (const uint4*)base;
    a.q[0] = p[0];                       // K +0..7   (16B)
    a.q[1] = p[2];                       // K +16..23 (16B)
}

// ---------------------------------------------------------------------------
// Kernel 0a: weight prep — transpose + bf16 convert (Wt[n*K+k] = W[k*N+n])
// grid (256, 4)  block 256
// ---------------------------------------------------------------------------
__global__ __launch_bounds__(256)
void prep_w_kernel(const float* __restrict__ Wq, const float* __restrict__ Wk,
                   const float* __restrict__ Wv, const float* __restrict__ W1,
                   __bf16* __restrict__ Wqt, __bf16* __restrict__ Wkt,
                   __bf16* __restrict__ Wvt, __bf16* __restrict__ W1t) {
    const int n = blockIdx.x, k = threadIdx.x, m = blockIdx.y;
    if (m == 0)      { if (k < 128) Wqt[n * 128 + k] = f2bf(Wq[k * 256 + n]); }
    else if (m == 1) Wkt[n * 256 + k] = f2bf(Wk[k * 256 + n]);
    else if (m == 2) Wvt[n * 256 + k] = f2bf(Wv[k * 256 + n]);
    else             W1t[n * 256 + k] = f2bf(W1[k * 256 + n]);
}

// ---------------------------------------------------------------------------
// Kernel 0b: Hobs fp32 -> bf16 (1,048,576 elems; 4 per thread, b64 stores)
// grid 1024, block 256
// ---------------------------------------------------------------------------
__global__ __launch_bounds__(256)
void prep_hobs_kernel(const float* __restrict__ Hobs, __bf16* __restrict__ Hb) {
    const int i = (blockIdx.x * 256 + threadIdx.x) * 4;
    const float4 f = *(const float4*)(Hobs + i);
    PK4 pk;
    pk.e[0] = f2bf(f.x); pk.e[1] = f2bf(f.y);
    pk.e[2] = f2bf(f.z); pk.e[3] = f2bf(f.w);
    *(uint2*)(Hb + i) = pk.d;
}

// ---------------------------------------------------------------------------
// Kernel 1: K/V projections (4096x256)@(256x256)+bias, bf16 in, bf16 out.
// One wave computes a 16x32 slab (A operand reused across 2 col tiles).
// K stored row-major; V stored TRANSPOSED Vt[b][feature][s].
// grid (32, 8, 2)  block 256
// ---------------------------------------------------------------------------
__global__ __launch_bounds__(256)
void kv_proj_kernel(const __bf16* __restrict__ Hb,
                    const __bf16* __restrict__ Wkt, const float* __restrict__ bk,
                    const __bf16* __restrict__ Wvt, const float* __restrict__ bv,
                    __bf16* __restrict__ Kbf, __bf16* __restrict__ Vt) {
    const int lane = threadIdx.x & 31, wave = threadIdx.x >> 5;
    const int hi = lane >> 4, lo = lane & 15;
    const int m0 = (blockIdx.x * 8 + wave) * 16;      // global row (b*512+s)
    const int n0 = blockIdx.y * 32;                   // feature slab (2 tiles)
    const bool isV = blockIdx.z != 0;
    const __bf16* Wt   = isV ? Wvt : Wkt;
    const float*  bias = isV ? bv : bk;

    const int mrow = m0 + lo;
    v8f acc0 = {}, acc1 = {};
#pragma unroll
    for (int kb = 0; kb < 256; kb += 32) {
        ABop a, b0, b1;
        load_bf16_runs(a,  Hb + mrow * 256 + kb + 8 * hi);
        load_bf16_runs(b0, Wt + (n0 + lo) * 256 + kb + 8 * hi);
        load_bf16_runs(b1, Wt + (n0 + 16 + lo) * 256 + kb + 8 * hi);
        acc0 = wmma_bf16(a.v, b0.v, acc0);
        acc1 = wmma_bf16(a.v, b1.v, acc1);
    }
#pragma unroll
    for (int half = 0; half < 2; ++half) {
        const v8f& acc = half ? acc1 : acc0;
        const int ncol = n0 + half * 16 + lo;
        const float bn = bias[ncol];
        if (!isV) {                                   // row-major store
#pragma unroll
            for (int r = 0; r < 8; ++r)
                Kbf[(m0 + r + 8 * hi) * 256 + ncol] = f2bf(acc[r] + bn);
        } else {                                      // transposed b128 store
            const int bt = m0 >> 9;                   // batch (512 rows each)
            const int sl = (m0 & 511) + 8 * hi;       // s within batch
            PK8 pk;
#pragma unroll
            for (int r = 0; r < 8; ++r) pk.e[r] = f2bf(acc[r] + bn);
            *(uint4*)(Vt + (size_t)bt * 131072 + (size_t)ncol * 512 + sl) = pk.q;
        }
    }
}

// ---------------------------------------------------------------------------
// Kernel 2: separable Q projection.
//   qa[bh,n] = A[bh,:64] @ Wq[0:64] ; qt[t,n] = [phi|X][t] @ Wq[64:128] + bq
// grid 64, block 256 -> 512 waves, one 16x16 tile each.
// ---------------------------------------------------------------------------
__global__ __launch_bounds__(256)
void q_parts_kernel(const float* __restrict__ Araw, const float* __restrict__ phi,
                    const float* __restrict__ Xt,   const __bf16* __restrict__ Wqt,
                    const float* __restrict__ bq,
                    float* __restrict__ qa, float* __restrict__ qt) {
    const int lane = threadIdx.x & 31, wave = threadIdx.x >> 5;
    const int hi = lane >> 4, lo = lane & 15;
    int g = blockIdx.x * 8 + wave;                    // 0..511
    const bool isQT = g >= 256;
    if (isQT) g -= 256;
    const int m0 = (g >> 4) * 16, n0 = (g & 15) * 16;
    const int m = m0 + lo, ncol = n0 + lo;

    v8f acc = {};
#pragma unroll
    for (int kb = 0; kb < 64; kb += 32) {
        ABop a, b;
        if (!isQT) {
            load_a_f32(a, Araw + m * 64 + kb + 8 * hi);
        } else {                                      // kb==0 -> phi, kb==32 -> X
            const float* src = (kb == 0) ? phi : Xt;
            load_a_f32(a, src + m * 32 + 8 * hi);
        }
        load_bf16_runs(b, Wqt + ncol * 128 + (isQT ? 64 : 0) + kb + 8 * hi);
        acc = wmma_bf16(a.v, b.v, acc);
    }
    const float bn = isQT ? bq[ncol] : 0.0f;
    float* out = isQT ? qt : qa;
#pragma unroll
    for (int r = 0; r < 8; ++r)
        out[(m0 + r + 8 * hi) * 256 + ncol] = acc[r] + bn;
}

// ---------------------------------------------------------------------------
// Kernel 3: attention. WG = (t-block of 32, head, b*H). 64 KB LDS score block;
// softmax -> bf16 in place; ctx = att @ V via WMMA (LDS b128 A, global b128 B).
// grid (8, 8, 256)  block 128 (4 waves)
// ---------------------------------------------------------------------------
#define TBLK 32
__global__ __launch_bounds__(128)
void attn_kernel(const float* __restrict__ qa, const float* __restrict__ qt,
                 const __bf16* __restrict__ Kbf, const __bf16* __restrict__ Vt,
                 __bf16* __restrict__ ctx) {
    __shared__ float sm[TBLK * 512];                  // 65536 B; reused as bf16 att
    const int lane = threadIdx.x & 31, wave = threadIdx.x >> 5;   // wave 0..3
    const int hi = lane >> 4, lo = lane & 15;
    const int tblk = blockIdx.x, head = blockIdx.y, bh = blockIdx.z;
    const int b = bh >> 5;                            // H == 32
    const int tbase = tblk * TBLK, fbase = head * 32;
    const size_t kbase  = (size_t)b * 512 * 256;      // K row-major base
    const size_t vtbase = (size_t)b * 131072;         // Vt [feature][s] base
    const float scale = 0.17677669529663687f;         // 1/sqrt(32) folded into Q

    // --- build scaled-bf16 Q operands (qa: per-(b,h); qt: per-target)
    ABop aop[2];
#pragma unroll
    for (int tm = 0; tm < 2; ++tm) {
        F16x va, vt;
        const float4* pa = (const float4*)(qa + bh * 256 + fbase + 8 * hi);
        const float4* pt =
            (const float4*)(qt + (tbase + tm * 16 + lo) * 256 + fbase + 8 * hi);
        va.f4[0] = pa[0]; va.f4[1] = pa[1]; va.f4[2] = pa[4]; va.f4[3] = pa[5];
        vt.f4[0] = pt[0]; vt.f4[1] = pt[1]; vt.f4[2] = pt[4]; vt.f4[3] = pt[5];
#pragma unroll
        for (int e = 0; e < 16; ++e)
            aop[tm].e[e] = f2bf((va.f[e] + vt.f[e]) * scale);
    }

    // --- scores S[t,s] = Q . K  (one WMMA per 16x16 tile, K=32 head dim)
    for (int tn = wave; tn < 32; tn += 4) {
        const int s = tn * 16 + lo;
        ABop bop;
        load_bf16_runs(bop, Kbf + kbase + (size_t)s * 256 + fbase + 8 * hi);
        if (tn + 4 < 32)                               // prefetch next s-tile
            __builtin_prefetch(Kbf + kbase + (size_t)(s + 64) * 256 + fbase, 0, 1);
#pragma unroll
        for (int tm = 0; tm < 2; ++tm) {
            v8f c = {};
            c = wmma_bf16(aop[tm].v, bop.v, c);
#pragma unroll
            for (int r = 0; r < 8; ++r)
                sm[(tm * 16 + r + 8 * hi) * 512 + tn * 16 + lo] = c[r];
        }
    }
    __syncthreads();

    // --- softmax over s per target row; convert to bf16 in place
    if (threadIdx.x < TBLK) {
        float* row = &sm[threadIdx.x * 512];
        float mx = -3.4e38f;
        for (int s = 0; s < 512; ++s) mx = fmaxf(mx, row[s]);
        float sum = 0.0f;
        for (int s = 0; s < 512; ++s) { float ev = __expf(row[s] - mx); row[s] = ev; sum += ev; }
        const float inv = 1.0f / sum;
        __bf16* rb = (__bf16*)row;
        for (int s = 0; s < 512; ++s) rb[s] = f2bf(row[s] * inv);
    }
    __syncthreads();

    // --- ctx[t,e] = sum_s att[t,s] * V[s,e] ; one 16x16 tile per wave
    {
        const int tm = wave >> 1, tn = wave & 1;
        const __bf16* arow = (const __bf16*)&sm[(tm * 16 + lo) * 512];
        const __bf16* vrow = Vt + vtbase + (size_t)(fbase + tn * 16 + lo) * 512;
        v8f c = {};
#pragma unroll 4
        for (int kb = 0; kb < 512; kb += 32) {
            ABop a, bop;
            load_bf16_runs(a,   arow + kb + 8 * hi);   // ds_load_b128 x2
            load_bf16_runs(bop, vrow + kb + 8 * hi);   // global_load_b128 x2
            c = wmma_bf16(a.v, bop.v, c);
        }
        const int col = fbase + tn * 16 + lo;
#pragma unroll
        for (int r = 0; r < 8; ++r) {
            const int t = tbase + tm * 16 + r + 8 * hi;
            ctx[((size_t)bh * 256 + t) * 256 + col] = f2bf(c[r]);
        }
    }
}

// ---------------------------------------------------------------------------
// Kernel 4: fused MLP head: y = relu(ctx@W1 + b1) @ W2 + b2
// grid 4096, block 256 (8 waves, 2 hidden-col tiles each).
// ---------------------------------------------------------------------------
__global__ __launch_bounds__(256)
void mlp_kernel(const __bf16* __restrict__ ctx, const __bf16* __restrict__ W1t,
                const float* __restrict__ b1,  const float* __restrict__ W2,
                const float* __restrict__ b2,  float* __restrict__ y) {
    __shared__ float part[16];
    const int lane = threadIdx.x & 31, wave = threadIdx.x >> 5;
    const int hi = lane >> 4, lo = lane & 15;
    const int m0 = blockIdx.x * 16;
    if (threadIdx.x < 16) part[threadIdx.x] = b2[0];
    __syncthreads();

    const __bf16* crow = ctx + (size_t)(m0 + lo) * 256;
    for (int ct = wave; ct < 16; ct += 8) {
        const int n0 = ct * 16, n = n0 + lo;
        const __bf16* wrow = W1t + (size_t)n * 256;
        v8f acc = {};
#pragma unroll
        for (int kb = 0; kb < 256; kb += 32) {
            ABop a, bop;
            load_bf16_runs(a,   crow + kb + 8 * hi);
            load_bf16_runs(bop, wrow + kb + 8 * hi);
            acc = wmma_bf16(a.v, bop.v, acc);
        }
        const float w2n = W2[n], b1n = b1[n];
#pragma unroll
        for (int r = 0; r < 8; ++r) {
            float p = fmaxf(acc[r] + b1n, 0.0f) * w2n;
#pragma unroll
            for (int off = 1; off < 16; off <<= 1)     // reduce across n lanes
                p += __shfl_xor(p, off, 32);
            if (lo == 0) atomicAdd(&part[r + 8 * hi], p);
        }
    }
    __syncthreads();
    if (threadIdx.x < 16) y[m0 + threadIdx.x] = part[threadIdx.x];
}

// ---------------------------------------------------------------------------
extern "C" void kernel_launch(void* const* d_in, const int* in_sizes, int n_in,
                              void* d_out, int out_size, void* d_ws, size_t ws_size,
                              hipStream_t stream) {
    const float* A    = (const float*)d_in[0];   // (8,32,64)
    const float* phi  = (const float*)d_in[1];   // (256,32)
    const float* Xt   = (const float*)d_in[2];   // (256,32)
    const float* Hobs = (const float*)d_in[3];   // (8,512,256)
    const float* Wq   = (const float*)d_in[4];   // (128,256)
    const float* bq   = (const float*)d_in[5];
    const float* Wk   = (const float*)d_in[6];   // (256,256)
    const float* bk   = (const float*)d_in[7];
    const float* Wv   = (const float*)d_in[8];
    const float* bv   = (const float*)d_in[9];
    const float* W1   = (const float*)d_in[10];  // (256,256)
    const float* b1   = (const float*)d_in[11];
    const float* W2   = (const float*)d_in[12];  // (256,1)
    const float* b2   = (const float*)d_in[13];
    float* y = (float*)d_out;                    // (8,32,256,1) fp32

    char* ws = (char*)d_ws;
    __bf16* Kbf = (__bf16*)(ws);                              //  2 MB row-major K
    __bf16* Vt  = (__bf16*)(ws + (2u << 20));                 //  2 MB transposed V
    float*  qa  = (float*)(ws + (4u << 20));                  // 256 KB
    float*  qt  = (float*)(ws + (4u << 20) + (256u << 10));   // 256 KB
    __bf16* Wqt = (__bf16*)(ws + (4u << 20) + (512u << 10));  //  64 KB
    __bf16* Wkt = (__bf16*)(ws + (4u << 20) + (576u << 10));  // 128 KB
    __bf16* Wvt = (__bf16*)(ws + (4u << 20) + (704u << 10));  // 128 KB
    __bf16* W1t = (__bf16*)(ws + (4u << 20) + (832u << 10));  // 128 KB
    __bf16* ctx = (__bf16*)(ws + (5u << 20));                 //  32 MB
    __bf16* Hb  = (__bf16*)(ws + (37u << 20));                //   2 MB bf16 Hobs

    prep_w_kernel<<<dim3(256, 4), dim3(256), 0, stream>>>(Wq, Wk, Wv, W1,
                                                          Wqt, Wkt, Wvt, W1t);
    prep_hobs_kernel<<<dim3(1024), dim3(256), 0, stream>>>(Hobs, Hb);
    kv_proj_kernel<<<dim3(32, 8, 2), dim3(256), 0, stream>>>(Hb, Wkt, bk, Wvt, bv,
                                                             Kbf, Vt);
    q_parts_kernel<<<dim3(64), dim3(256), 0, stream>>>(A, phi, Xt, Wqt, bq, qa, qt);
    attn_kernel<<<dim3(8, 8, 256), dim3(128), 0, stream>>>(qa, qt, Kbf, Vt, ctx);
    mlp_kernel<<<dim3(4096), dim3(256), 0, stream>>>(ctx, W1t, b1, W2, b2, y);
}
